// Generator_1932735283569
// MI455X (gfx1250) — compile-verified, matrix operand order checked by reading, not run on hardware
//
#include <hip/hip_runtime.h>
#include <hip/hip_bf16.h>

typedef __bf16 bf16_t;
typedef __attribute__((ext_vector_type(16))) __bf16 v16bf;
typedef __attribute__((ext_vector_type(8)))  float  v8f;
typedef int v4i32 __attribute__((vector_size(16)));

#define B_SZ   512
#define T_SZ   256
#define D_SZ   128
#define H_SZ   512
#define H4     2048
#define K0TOT  640    // D + H   (layer-0 fused K)
#define K1TOT  1024   // H + H   (layer-1 fused K)
#define S0     656    // LDS stride (elements) for in0, 656*2B = 41*32B
#define S1     1040   // LDS stride for in1, 1040*2B = 65*32B
#define SX     132    // LDS stride for x tile (floats)
#define NPRED  (B_SZ*(T_SZ-1)*D_SZ)

#if defined(__has_builtin)
#  if __has_builtin(__builtin_amdgcn_global_load_async_to_lds_b128) && \
      __has_builtin(__builtin_amdgcn_s_wait_asynccnt)
#    define HAS_ASYNC_LDS 1
#  endif
#endif
#ifndef HAS_ASYNC_LDS
#  define HAS_ASYNC_LDS 0
#endif

__device__ __forceinline__ v16bf ld16(const bf16_t* p) { return *(const v16bf*)p; }
__device__ __forceinline__ float sigmoidf_(float x) { return 1.0f / (1.0f + __expf(-x)); }
__device__ __forceinline__ float tanhf_(float x) { float e = __expf(2.0f * x); return (e - 1.0f) / (e + 1.0f); }

#define WMMA_BF16(A, Bm, C) \
    __builtin_amdgcn_wmma_f32_16x16x32_bf16(false, (A), false, (Bm), (short)0, (C), false, false)

#if HAS_ASYNC_LDS
__device__ __forceinline__ void async_ld16B(const void* g, void* l) {
    __builtin_amdgcn_global_load_async_to_lds_b128(
        (__attribute__((address_space(1))) v4i32*)g,
        (__attribute__((address_space(3))) v4i32*)l, 0, 0);
}
#endif

// ---- weight prep: f32 -> bf16, transposed to [N][K] so B-tiles load contiguously ----
__global__ void prep_w0(const float* __restrict__ k0, const float* __restrict__ r0, bf16_t* __restrict__ o) {
    int idx = blockIdx.x * 256 + threadIdx.x;
    if (idx >= H4 * K0TOT) return;
    int n = idx / K0TOT, k = idx % K0TOT;
    float v = (k < D_SZ) ? k0[(size_t)k * H4 + n] : r0[(size_t)(k - D_SZ) * H4 + n];
    o[idx] = (bf16_t)v;
}
__global__ void prep_w1(const float* __restrict__ k1, const float* __restrict__ r1, bf16_t* __restrict__ o) {
    int idx = blockIdx.x * 256 + threadIdx.x;
    if (idx >= H4 * K1TOT) return;
    int n = idx / K1TOT, k = idx % K1TOT;
    float v = (k < H_SZ) ? k1[(size_t)k * H4 + n] : r1[(size_t)(k - H_SZ) * H4 + n];
    o[idx] = (bf16_t)v;
}
__global__ void prep_wt(const float* __restrict__ W, bf16_t* __restrict__ o) {
    int idx = blockIdx.x * 256 + threadIdx.x;
    if (idx >= D_SZ * H_SZ) return;
    int d = idx / H_SZ, h = idx % H_SZ;
    o[idx] = (bf16_t)W[(size_t)h * D_SZ + d];
}

// ---- persistent 2-layer LSTM with imputation; one WG (32 waves) per 16 batch rows ----
__launch_bounds__(1024, 1)
__global__ void lstm_impute_persistent(
    const float* __restrict__ x,
    const float* __restrict__ b0, const float* __restrict__ b1,
    const float* __restrict__ bout,
    const bf16_t* __restrict__ w0t,   // [2048][640]
    const bf16_t* __restrict__ w1t,   // [2048][1024]
    const bf16_t* __restrict__ wt,    // [128][512]
    float* __restrict__ out)
{
    __shared__ __attribute__((aligned(32))) bf16_t in0[16 * S0]; // [x(128) | h0(512)] bf16
    __shared__ __attribute__((aligned(32))) bf16_t in1[16 * S1]; // [h0(512) | h1(512)] bf16
    __shared__ __attribute__((aligned(16))) float  xs[16 * SX];  // x_t tile f32

    const int tid   = threadIdx.x;
    const int lane  = tid & 31;
    const int wave  = tid >> 5;       // 0..31
    const int rbase = blockIdx.x * 16;

    const int l15   = lane & 15;
    const int khalf = (lane >> 4) << 4;   // K-half select for A/B tiles
    const int mrow  = lane >> 4;          // C/D row-half

    for (int i = tid; i < 16 * S0; i += 1024) in0[i] = (bf16_t)0.0f;
    for (int i = tid; i < 16 * S1; i += 1024) in1[i] = (bf16_t)0.0f;

    // this lane's output column (each wave owns 16 h-columns, all 4 gates)
    const int col = wave * 16 + l15;

    // hoisted loop-invariant scalars
    const float bi0 = b0[col],            bf0_ = b0[H_SZ + col];
    const float bg0 = b0[2 * H_SZ + col], bo0  = b0[3 * H_SZ + col];
    const float bi1 = b1[col],            bf1_ = b1[H_SZ + col];
    const float bg1 = b1[2 * H_SZ + col], bo1  = b1[3 * H_SZ + col];
    const float bd  = (wave < 8) ? bout[col] : 0.0f;

    // hoisted weight-row base pointers (row = g*H + col, contiguous K)
    const bf16_t* w0p0 = w0t + (size_t)(0 * H_SZ + col) * K0TOT + khalf;
    const bf16_t* w0p1 = w0t + (size_t)(1 * H_SZ + col) * K0TOT + khalf;
    const bf16_t* w0p2 = w0t + (size_t)(2 * H_SZ + col) * K0TOT + khalf;
    const bf16_t* w0p3 = w0t + (size_t)(3 * H_SZ + col) * K0TOT + khalf;
    const bf16_t* w1p0 = w1t + (size_t)(0 * H_SZ + col) * K1TOT + khalf;
    const bf16_t* w1p1 = w1t + (size_t)(1 * H_SZ + col) * K1TOT + khalf;
    const bf16_t* w1p2 = w1t + (size_t)(2 * H_SZ + col) * K1TOT + khalf;
    const bf16_t* w1p3 = w1t + (size_t)(3 * H_SZ + col) * K1TOT + khalf;
    const bf16_t* wtp  = (wave < 8) ? (wt + (size_t)col * H_SZ + khalf) : wt;

    // hoisted LDS A-tile base pointers
    const bf16_t* ain0  = &in0[l15 * S0 + khalf];
    const bf16_t* ain1  = &in1[l15 * S1 + khalf];
    const bf16_t* ah1   = &in1[l15 * S1 + H_SZ + khalf];

    const v8f zero8 = {0.f, 0.f, 0.f, 0.f, 0.f, 0.f, 0.f, 0.f};
    v8f c0 = zero8, c1 = zero8;
    v8f accP = zero8;

    float* out_pred = out;
    float* out_last = out + (size_t)NPRED;

    __syncthreads();

    for (int t = 0; t < T_SZ; ++t) {
        // ---- Phase A: imputation GEMM (waves 0-7) | x_t staging (waves 8-15) ----
        if (wave < 8) {
            accP = zero8;
            if (t > 0) {
                for (int kk = 0; kk < H_SZ; kk += 32) {
                    v16bf a = ld16(ah1 + kk);       // h1 prev, A-tile
                    v16bf b = ld16(wtp + kk);       // W^T rows
                    accP = WMMA_BF16(a, b, accP);
                }
            }
        } else if (wave < 16) {
            const int tid2 = tid - 256;  // 0..255
            const float* xg = x + (size_t)rbase * T_SZ * D_SZ + (size_t)t * D_SZ;
#if HAS_ASYNC_LDS
            #pragma unroll
            for (int j = 0; j < 2; ++j) {
                int c = tid2 + j * 256;       // 0..511 b128 chunks (16 rows x 32 chunks)
                int m = c >> 5;
                int f = (c & 31) * 4;
                async_ld16B(xg + (size_t)m * T_SZ * D_SZ + f, &xs[m * SX + f]);
            }
            __builtin_amdgcn_s_wait_asynccnt(0);
#else
            for (int j = tid2; j < 16 * D_SZ / 4; j += 256) {
                int m = j / 32, d = (j % 32) * 4;
                float4 v = *(const float4*)(xg + (size_t)m * T_SZ * D_SZ + d);
                *(float4*)&xs[m * SX + d] = v;
            }
#endif
            if (t + 1 < T_SZ) { // prefetch next step's x tile into cache
                __builtin_prefetch(xg + (size_t)(tid2 >> 4) * T_SZ * D_SZ + D_SZ + (tid2 & 15) * 8, 0, 0);
            }
        }
        __syncthreads();

        // ---- Phase B: fuse bias + missing-mask; emit prediction; build layer-0 input ----
        if (wave < 8) {
            #pragma unroll
            for (int i = 0; i < 8; ++i) {
                const int m = mrow * 8 + i;
                float xv = xs[m * SX + col];
                float pv = accP[i] + bd;
                float cv = ((t > 0) && (xv == 128.0f)) ? pv : xv;
                in0[m * S0 + col] = (bf16_t)cv;
                if (t > 0)
                    out_pred[((size_t)(rbase + m) * (T_SZ - 1) + (t - 1)) * D_SZ + col] = pv;
            }
        }
        __syncthreads();

        // ---- Phase C: layer-0 fused GEMM z = [x|h0] @ [k0;r0], K = 640 ----
        v8f z0 = zero8, z1 = zero8, z2 = zero8, z3 = zero8;
        for (int kk = 0; kk < K0TOT; kk += 32) {
            v16bf a  = ld16(ain0 + kk);
            v16bf q0 = ld16(w0p0 + kk);
            v16bf q1 = ld16(w0p1 + kk);
            v16bf q2 = ld16(w0p2 + kk);
            v16bf q3 = ld16(w0p3 + kk);
            z0 = WMMA_BF16(a, q0, z0);
            z1 = WMMA_BF16(a, q1, z1);
            z2 = WMMA_BF16(a, q2, z2);
            z3 = WMMA_BF16(a, q3, z3);
        }
        __syncthreads();

        // ---- Phase C2: layer-0 gates, update c0, write h0 (bf16) ----
        #pragma unroll
        for (int i = 0; i < 8; ++i) {
            float iv = sigmoidf_(z0[i] + bi0);
            float fv = sigmoidf_(z1[i] + bf0_);
            float gv = tanhf_  (z2[i] + bg0);
            float ov = sigmoidf_(z3[i] + bo0);
            float cv = fv * c0[i] + iv * gv;
            c0[i] = cv;
            float hv = ov * tanhf_(cv);
            const int m = mrow * 8 + i;
            bf16_t hb = (bf16_t)hv;
            in0[m * S0 + D_SZ + col] = hb;  // next step's layer-0 input
            in1[m * S1 + col]        = hb;  // layer-1 input (this step)
        }
        __syncthreads();

        // ---- Phase D: layer-1 fused GEMM z = [h0|h1] @ [k1;r1], K = 1024 ----
        z0 = zero8; z1 = zero8; z2 = zero8; z3 = zero8;
        for (int kk = 0; kk < K1TOT; kk += 32) {
            v16bf a  = ld16(ain1 + kk);
            v16bf q0 = ld16(w1p0 + kk);
            v16bf q1 = ld16(w1p1 + kk);
            v16bf q2 = ld16(w1p2 + kk);
            v16bf q3 = ld16(w1p3 + kk);
            z0 = WMMA_BF16(a, q0, z0);
            z1 = WMMA_BF16(a, q1, z1);
            z2 = WMMA_BF16(a, q2, z2);
            z3 = WMMA_BF16(a, q3, z3);
        }
        __syncthreads();

        // ---- Phase E: layer-1 gates, update c1, write h1; final-step output ----
        #pragma unroll
        for (int i = 0; i < 8; ++i) {
            float iv = sigmoidf_(z0[i] + bi1);
            float fv = sigmoidf_(z1[i] + bf1_);
            float gv = tanhf_  (z2[i] + bg1);
            float ov = sigmoidf_(z3[i] + bo1);
            float cv = fv * c1[i] + iv * gv;
            c1[i] = cv;
            float hv = ov * tanhf_(cv);
            const int m = mrow * 8 + i;
            in1[m * S1 + H_SZ + col] = (bf16_t)hv;
            if (t == T_SZ - 1)
                out_last[(size_t)(rbase + m) * H_SZ + col] = hv;
        }
        __syncthreads();
    }
}

extern "C" void kernel_launch(void* const* d_in, const int* in_sizes, int n_in,
                              void* d_out, int out_size, void* d_ws, size_t ws_size,
                              hipStream_t stream) {
    (void)in_sizes; (void)n_in; (void)out_size; (void)ws_size;
    const float* x  = (const float*)d_in[0];
    const float* k0 = (const float*)d_in[1];
    const float* r0 = (const float*)d_in[2];
    const float* b0 = (const float*)d_in[3];
    const float* k1 = (const float*)d_in[4];
    const float* r1 = (const float*)d_in[5];
    const float* b1 = (const float*)d_in[6];
    const float* W  = (const float*)d_in[7];
    const float* bo = (const float*)d_in[8];

    bf16_t* w0t = (bf16_t*)d_ws;
    bf16_t* w1t = (bf16_t*)((char*)d_ws + (size_t)H4 * K0TOT * sizeof(bf16_t));
    bf16_t* wt  = (bf16_t*)((char*)d_ws + (size_t)H4 * K0TOT * sizeof(bf16_t)
                                        + (size_t)H4 * K1TOT * sizeof(bf16_t));

    prep_w0<<<(H4 * K0TOT + 255) / 256, 256, 0, stream>>>(k0, r0, w0t);
    prep_w1<<<(H4 * K1TOT + 255) / 256, 256, 0, stream>>>(k1, r1, w1t);
    prep_wt<<<(D_SZ * H_SZ + 255) / 256, 256, 0, stream>>>(W, wt);

    lstm_impute_persistent<<<B_SZ / 16, 1024, 0, stream>>>(
        x, b0, b1, bo, w0t, w1t, wt, (float*)d_out);
}
